// PointerNetwork_30485677867997
// MI455X (gfx1250) — compile-verified, matrix operand order checked by reading, not run on hardware
//
#include <hip/hip_runtime.h>
#include <hip/hip_bf16.h>

// ---------------------------------------------------------------------------
// Types for CDNA5 WMMA (wave32): v16bf = 8 VGPRs of bf16, v8f = 8 VGPRs f32.
// ---------------------------------------------------------------------------
typedef __bf16 bf16_t;
typedef bf16_t v16bf __attribute__((ext_vector_type(16)));
typedef float  v8f   __attribute__((ext_vector_type(8)));
typedef unsigned int u32x4 __attribute__((ext_vector_type(4)));
typedef int          i32x4 __attribute__((ext_vector_type(4)));
typedef int          i32x8 __attribute__((ext_vector_type(8)));

#define N_GRAPHS  64
#define MAX_NODES 256
#define DIM       1024
#define N_PTR     512

#if defined(__HIP_DEVICE_COMPILE__) && defined(__has_builtin)
#  if __has_builtin(__builtin_amdgcn_tensor_load_to_lds) && \
      __has_builtin(__builtin_amdgcn_s_wait_tensorcnt)
#    define HAVE_TDM 1
#  endif
#endif
#ifndef HAVE_TDM
#  define HAVE_TDM 0
#endif

// ---------------------------------------------------------------------------
// One-shot fp32 -> bf16 conversion (streaming, 8 elements / thread).
// Removes all v_cvt work and half the load bytes from the GEMM hot loop;
// inputs total ~70 MB so this pass costs ~3 us at 23.3 TB/s.
// ---------------------------------------------------------------------------
__global__ void __launch_bounds__(256)
cvt_f32_bf16_kernel(const float* __restrict__ in, bf16_t* __restrict__ out) {
  const size_t i = ((size_t)blockIdx.x * 256 + threadIdx.x) * 8;
  float4 a = *(const float4*)(in + i);
  float4 b = *(const float4*)(in + i + 4);
  union { bf16_t h[8]; uint4 q; } u;
  u.h[0] = (bf16_t)a.x; u.h[1] = (bf16_t)a.y;
  u.h[2] = (bf16_t)a.z; u.h[3] = (bf16_t)a.w;
  u.h[4] = (bf16_t)b.x; u.h[5] = (bf16_t)b.y;
  u.h[6] = (bf16_t)b.z; u.h[7] = (bf16_t)b.w;
  *(uint4*)(out + i) = u.q;
}

// ---------------------------------------------------------------------------
// Load one 16x32 bf16 fragment for v_wmma_f32_16x16x32_bf16 from a row-major
// bf16 matrix (ld = DIM). Per ISA 7.12.2 (16-bit A 16x32):
//   lane = r + 16*h ; elems 0..7 -> K = k0+8h+j ; elems 8..15 -> K = k0+16+8h+j
// Two 16-byte (b128) loads per lane, K-contiguous for both A rows and W rows.
// ---------------------------------------------------------------------------
__device__ __forceinline__ v16bf load_frag(const bf16_t* __restrict__ base,
                                           int row0, int k0, int lane) {
  const int r = lane & 15;
  const int h = lane >> 4;
  const bf16_t* p = base + (size_t)(row0 + r) * DIM + (k0 + h * 8);
  union { v16bf v; uint4 q[2]; } u;
  u.q[0] = *(const uint4*)(p);
  u.q[1] = *(const uint4*)(p + 16);
  return u.v;
}

// ---------------------------------------------------------------------------
// C[m,n] = sum_d A[m,d] * W[n,d] + bias[n]   (A, W bf16; C fp32)
// Block = 256 threads = 8 waves as 4(M) x 2(N); each wave owns a 32x64 tile
// (2x4 WMMA fragments, 8 WMMA per 32-wide k-chunk, A frags reused 4x).
// Block tile 128(M) x 128(N). Grid: (1024/128, M/128).
// ---------------------------------------------------------------------------
__global__ void __launch_bounds__(256)
gemm_bias_bf16_kernel(const bf16_t* __restrict__ A, const bf16_t* __restrict__ W,
                      const float* __restrict__ bias, float* __restrict__ C) {
  const int lane = threadIdx.x & 31;
  const int wave = threadIdx.x >> 5;
  const int wm = wave & 3;   // 0..3
  const int wn = wave >> 2;  // 0..1
  const int row0 = blockIdx.y * 128 + wm * 32;
  const int col0 = blockIdx.x * 128 + wn * 64;

  v8f acc[2][4] = {};

  for (int k0 = 0; k0 < DIM; k0 += 32) {
    v16bf a0 = load_frag(A, row0,      k0, lane);
    v16bf a1 = load_frag(A, row0 + 16, k0, lane);
    v16bf b0 = load_frag(W, col0,      k0, lane);
    v16bf b1 = load_frag(W, col0 + 16, k0, lane);
    v16bf b2 = load_frag(W, col0 + 32, k0, lane);
    v16bf b3 = load_frag(W, col0 + 48, k0, lane);
    acc[0][0] = __builtin_amdgcn_wmma_f32_16x16x32_bf16(false, a0, false, b0,
                                                        (short)0, acc[0][0], false, false);
    acc[0][1] = __builtin_amdgcn_wmma_f32_16x16x32_bf16(false, a0, false, b1,
                                                        (short)0, acc[0][1], false, false);
    acc[0][2] = __builtin_amdgcn_wmma_f32_16x16x32_bf16(false, a0, false, b2,
                                                        (short)0, acc[0][2], false, false);
    acc[0][3] = __builtin_amdgcn_wmma_f32_16x16x32_bf16(false, a0, false, b3,
                                                        (short)0, acc[0][3], false, false);
    acc[1][0] = __builtin_amdgcn_wmma_f32_16x16x32_bf16(false, a1, false, b0,
                                                        (short)0, acc[1][0], false, false);
    acc[1][1] = __builtin_amdgcn_wmma_f32_16x16x32_bf16(false, a1, false, b1,
                                                        (short)0, acc[1][1], false, false);
    acc[1][2] = __builtin_amdgcn_wmma_f32_16x16x32_bf16(false, a1, false, b2,
                                                        (short)0, acc[1][2], false, false);
    acc[1][3] = __builtin_amdgcn_wmma_f32_16x16x32_bf16(false, a1, false, b3,
                                                        (short)0, acc[1][3], false, false);
  }

  // C/D layout: lane = n + 16*(m/8), VGPR i = m%8.
  const int h  = lane >> 4;
  const int nl = lane & 15;
  float bn[4];
#pragma unroll
  for (int j = 0; j < 4; ++j) bn[j] = bias[col0 + j * 16 + nl];
#pragma unroll
  for (int i2 = 0; i2 < 2; ++i2) {
#pragma unroll
    for (int j = 0; j < 4; ++j) {
      const int n = col0 + j * 16 + nl;
#pragma unroll
      for (int i = 0; i < 8; ++i) {
        const int m = row0 + i2 * 16 + i + 8 * h;
        C[(size_t)m * DIM + n] = acc[i2][j][i] + bn[j];
      }
    }
  }
}

// ---------------------------------------------------------------------------
// scores[p,n] = sum_d relu(X[g_p*256+n, d] + Y[p, d]) * w_s[d], -inf padded.
// Graph-major: block = (graph g, 32-node chunk). The contiguous 32x1024 fp32
// slab (128 KB) is DMA'd into LDS by the Tensor Data Mover (wave 0 issues one
// TENSOR_LOAD_TO_LDS; TDM ignores EXEC and signals TENSORcnt once), then
// reused for every pointer with graph_ids[p] == g. X leaves HBM exactly once.
// ---------------------------------------------------------------------------
__global__ void __launch_bounds__(256)
scores_kernel(const float* __restrict__ X,            // (64*256, 1024)
              const float* __restrict__ Y,            // (512, 1024)
              const int* __restrict__ graph_ids,      // (512)
              const int* __restrict__ graph_sizes,    // (64)
              const float* __restrict__ w_s,          // (1024)
              float* __restrict__ out) {              // (512, 256)
  const int g     = blockIdx.x;  // 0..63
  const int chunk = blockIdx.y;  // 0..7
  const int tid   = threadIdx.x;
  const int lane  = tid & 31;
  const int wave  = tid >> 5;

  __shared__ int s_cnt;
  __shared__ int s_list[N_PTR];
  extern __shared__ float s_x[];  // 32 * 1024 floats = 128 KB

  if (tid == 0) s_cnt = 0;
  __syncthreads();

  for (int p = tid; p < N_PTR; p += 256) {
    if (graph_ids[p] == g) {
      int i = atomicAdd(&s_cnt, 1);
      s_list[i] = p;
    }
  }

  const float* slab = X + ((size_t)g * MAX_NODES + chunk * 32) * DIM;

#if HAVE_TDM
  if (wave == 0) {
    const unsigned long long ga = (unsigned long long)(uintptr_t)slab;
    const unsigned lds_addr = (unsigned)(uintptr_t)(void*)s_x;  // addr[31:0] = LDS offset
    // D# group 0: count=1 | lds_addr | global_addr[56:0] | type=2
    u32x4 g0;
    g0[0] = 1u;
    g0[1] = lds_addr;
    g0[2] = (unsigned)(ga & 0xffffffffu);
    g0[3] = (unsigned)((ga >> 32) & 0x01ffffffu) | (2u << 30);
    // D# group 1: data_size=2 (4B); tensor 1024 x 32, stride 1024;
    // tile 1024(x) x 32(y)
    i32x8 g1;
    g1[0] = (int)(2u << 16);                 // data_size = 4 bytes
    g1[1] = (int)((1024u & 0xffffu) << 16);  // tensor_dim0 lo16 in [63:48]
    g1[2] = (int)((1024u >> 16) | (32u << 16));   // dim0 hi16 | tensor_dim1 lo16
    g1[3] = (int)((1024u & 0xffffu) << 16);  // tensor_dim1 hi16=0 | tile_dim0=1024
    g1[4] = (int)32;                         // tile_dim1 = 32 | tile_dim2 = 0
    g1[5] = (int)1024;                       // tensor_dim0_stride lo32
    g1[6] = 0;                               // stride0 hi16 | stride1 lo16
    g1[7] = 0;
    i32x4 gz = {0, 0, 0, 0};
#if __clang_major__ >= 23
    i32x8 gz8 = {0, 0, 0, 0, 0, 0, 0, 0};
    __builtin_amdgcn_tensor_load_to_lds(g0, g1, gz, gz, gz8, 0);
#else
    __builtin_amdgcn_tensor_load_to_lds(g0, g1, gz, gz, 0);
#endif
    __builtin_amdgcn_s_wait_tensorcnt(0);
  }
#else
  {
    const float4* src = (const float4*)slab;
    float4* dst = (float4*)s_x;
    for (int i = tid; i < 32 * (DIM / 4); i += 256) dst[i] = src[i];
  }
#endif

  // Cache w_s in registers: lane l holds d = i*128 + 4*l .. +3 for i=0..7.
  float4 wsr[8];
  const float4* ws4 = (const float4*)w_s;
#pragma unroll
  for (int i = 0; i < 8; ++i) wsr[i] = ws4[i * 32 + lane];

  __syncthreads();

  const int gsz   = graph_sizes[g];
  const int nptrs = s_cnt;

  for (int j = 0; j < nptrs; ++j) {
    const int p = s_list[j];
    float4 yr[8];
    const float4* y4 = (const float4*)(Y + (size_t)p * DIM);
#pragma unroll
    for (int i = 0; i < 8; ++i) yr[i] = y4[i * 32 + lane];

#pragma unroll
    for (int nn = 0; nn < 4; ++nn) {
      const int n = wave + nn * 8;  // 0..31 within chunk
      const float4* x4 = (const float4*)(s_x + (size_t)n * DIM);
      float acc = 0.f;
#pragma unroll
      for (int i = 0; i < 8; ++i) {
        float4 xv = x4[i * 32 + lane];
        float r0 = fmaxf(xv.x + yr[i].x, 0.f);
        float r1 = fmaxf(xv.y + yr[i].y, 0.f);
        float r2 = fmaxf(xv.z + yr[i].z, 0.f);
        float r3 = fmaxf(xv.w + yr[i].w, 0.f);
        acc = fmaf(r0, wsr[i].x, acc);
        acc = fmaf(r1, wsr[i].y, acc);
        acc = fmaf(r2, wsr[i].z, acc);
        acc = fmaf(r3, wsr[i].w, acc);
      }
#pragma unroll
      for (int off = 16; off > 0; off >>= 1) acc += __shfl_xor(acc, off, 32);

      if (lane == 0) {
        const int node = chunk * 32 + n;
        out[(size_t)p * MAX_NODES + node] =
            (node >= gsz) ? -__builtin_inff() : acc;
      }
    }
  }
}

// ---------------------------------------------------------------------------
// Inputs: 0 graph_sizes 1 embeddings 2 predicted_embs 3 target_ptr_ids(unused)
//         4 graph_ids 5 Wx 6 bx 7 Wy 8 by 9 ws
// Workspace: Ebf(32MB) Pbf(1MB) Wxbf(2MB) Wybf(2MB) X(64MB f32) Y(2MB f32)
// ---------------------------------------------------------------------------
extern "C" void kernel_launch(void* const* d_in, const int* in_sizes, int n_in,
                              void* d_out, int out_size, void* d_ws, size_t ws_size,
                              hipStream_t stream) {
  (void)in_sizes; (void)n_in; (void)out_size; (void)ws_size;
  const int*   graph_sizes = (const int*)d_in[0];
  const float* embeddings  = (const float*)d_in[1];
  const float* predicted   = (const float*)d_in[2];
  const int*   graph_ids   = (const int*)d_in[4];
  const float* Wx          = (const float*)d_in[5];
  const float* bx          = (const float*)d_in[6];
  const float* Wy          = (const float*)d_in[7];
  const float* by          = (const float*)d_in[8];
  const float* w_s         = (const float*)d_in[9];

  const size_t nE = (size_t)N_GRAPHS * MAX_NODES * DIM;  // 16,777,216
  const size_t nP = (size_t)N_PTR * DIM;                 // 524,288
  const size_t nW = (size_t)DIM * DIM;                   // 1,048,576

  bf16_t* Ebf  = (bf16_t*)d_ws;
  bf16_t* Pbf  = Ebf + nE;
  bf16_t* Wxbf = Pbf + nP;
  bf16_t* Wybf = Wxbf + nW;
  float*  X    = (float*)(Wybf + nW);        // 16384 x 1024
  float*  Yp   = X + nE;                     // 512 x 1024
  float*  out  = (float*)d_out;              // 512 x 256

  dim3 blk(256);
  // fp32 -> bf16 streaming conversions (grid = elems / (8*256), all exact).
  cvt_f32_bf16_kernel<<<dim3((unsigned)(nE / 2048)), blk, 0, stream>>>(embeddings, Ebf);
  cvt_f32_bf16_kernel<<<dim3((unsigned)(nP / 2048)), blk, 0, stream>>>(predicted, Pbf);
  cvt_f32_bf16_kernel<<<dim3((unsigned)(nW / 2048)), blk, 0, stream>>>(Wx, Wxbf);
  cvt_f32_bf16_kernel<<<dim3((unsigned)(nW / 2048)), blk, 0, stream>>>(Wy, Wybf);

  // X = embeddings @ Wx^T + bx : grid (1024/128, 16384/128)
  gemm_bias_bf16_kernel<<<dim3(DIM / 128, (N_GRAPHS * MAX_NODES) / 128), blk, 0,
                          stream>>>(Ebf, Wxbf, bx, X);
  // Y = predicted @ Wy^T + by : grid (1024/128, 512/128)
  gemm_bias_bf16_kernel<<<dim3(DIM / 128, N_PTR / 128), blk, 0, stream>>>(
      Pbf, Wybf, by, Yp);

  // Fused relu-score + mask; TDM stages 128 KB X slabs into LDS.
  scores_kernel<<<dim3(N_GRAPHS, MAX_NODES / 32), blk, 32 * DIM * sizeof(float),
                  stream>>>(X, Yp, graph_ids, graph_sizes, w_s, out);
}
